// PointNetSetAbstractionMsg_60129542144285
// MI455X (gfx1250) — compile-verified, matrix operand order checked by reading, not run on hardware
//
#include <hip/hip_runtime.h>
#include <hip/hip_bf16.h>

#define BATCH 16
#define NPTS  4096
#define NPT   512

typedef __attribute__((ext_vector_type(16))) _Float16 v16h;
typedef __attribute__((ext_vector_type(8)))  _Float16 v8h;
typedef __attribute__((ext_vector_type(8)))  float    v8f;
typedef __attribute__((ext_vector_type(4)))  int      i4v;

union V16H { v16h v; v8h half8[2]; };
union H2   { _Float16 h[2]; unsigned u; };

#ifndef HAVE_ASYNC_LDS
#if defined(__has_builtin)
#if __has_builtin(__builtin_amdgcn_global_load_async_to_lds_b128) && \
    __has_builtin(__builtin_amdgcn_s_wait_asynccnt)
#define HAVE_ASYNC_LDS 1
#endif
#endif
#endif
#ifndef HAVE_ASYNC_LDS
#define HAVE_ASYNC_LDS 0
#endif

// ---------------------------------------------------------------------------
// Kernel 1: farthest point sampling. One workgroup per batch; dmin in LDS.
// ---------------------------------------------------------------------------
__global__ __launch_bounds__(256) void fps_kernel(const float* __restrict__ xyz,
                                                  float* __restrict__ nxyz,
                                                  float* __restrict__ out_xyz) {
  __shared__ float dmin[NPTS];
  __shared__ float rval[256];
  __shared__ int   ridx[256];
  __shared__ int   s_far;
  const int b = blockIdx.x;
  const int tid = threadIdx.x;
  const float* X = xyz + (size_t)b * 3 * NPTS;
  for (int j = tid; j < NPTS; j += 256) dmin[j] = 1e10f;
  if (tid == 0) s_far = 0;
  __syncthreads();
  for (int i = 0; i < NPT; ++i) {
    const int far = s_far;
    const float cx = X[far], cy = X[NPTS + far], cz = X[2 * NPTS + far];
    if (tid == 0) {
      float* w = nxyz + ((size_t)b * NPT + i) * 3;
      w[0] = cx; w[1] = cy; w[2] = cz;
      out_xyz[((size_t)b * 3 + 0) * NPT + i] = cx;
      out_xyz[((size_t)b * 3 + 1) * NPT + i] = cy;
      out_xyz[((size_t)b * 3 + 2) * NPT + i] = cz;
    }
    float best = -1.0f; int bi = 0;
    for (int j = tid; j < NPTS; j += 256) {
      float dx = X[j] - cx, dy = X[NPTS + j] - cy, dz = X[2 * NPTS + j] - cz;
      float d = dx * dx + dy * dy + dz * dz;
      float dm = fminf(dmin[j], d);
      dmin[j] = dm;
      if (dm > best) { best = dm; bi = j; }
    }
    rval[tid] = best; ridx[tid] = bi;
    __syncthreads();
    for (int off = 128; off > 0; off >>= 1) {
      if (tid < off) {
        float v2 = rval[tid + off]; int i2 = ridx[tid + off];
        if (v2 > rval[tid] || (v2 == rval[tid] && i2 < ridx[tid])) {
          rval[tid] = v2; ridx[tid] = i2;
        }
      }
      __syncthreads();
    }
    if (tid == 0) s_far = ridx[0];
    __syncthreads();
  }
}

// ---------------------------------------------------------------------------
// Kernel 2: ball query, all 3 radii in one pass. One wave32 per centroid.
// ---------------------------------------------------------------------------
__global__ __launch_bounds__(256) void ballquery_kernel(const float* __restrict__ xyz,
                                                        const float* __restrict__ nxyz,
                                                        int* __restrict__ o0g,
                                                        int* __restrict__ o1g,
                                                        int* __restrict__ o2g) {
  const int wv = threadIdx.x >> 5;
  const int lane = threadIdx.x & 31;
  const int cid = blockIdx.x * 8 + wv;
  const int b = cid >> 9;
  const float* X = xyz + (size_t)b * 3 * NPTS;
  const float cx = nxyz[(size_t)cid * 3 + 0];
  const float cy = nxyz[(size_t)cid * 3 + 1];
  const float cz = nxyz[(size_t)cid * 3 + 2];
  int* o0 = o0g + (size_t)cid * 16;
  int* o1 = o1g + (size_t)cid * 32;
  int* o2 = o2g + (size_t)cid * 128;
  int c0 = 0, c1 = 0, c2 = 0, f0 = 0, f1 = 0, f2 = 0;
  const unsigned below = (1u << lane) - 1u;
  for (int base = 0; base < NPTS; base += 32) {
    const int j = base + lane;
    float dx = X[j] - cx, dy = X[NPTS + j] - cy, dz = X[2 * NPTS + j] - cz;
    float d2 = dx * dx + dy * dy + dz * dz;
    unsigned m0 = (unsigned)__ballot(d2 <= 0.01f);
    unsigned m1 = (unsigned)__ballot(d2 <= 0.04f);
    unsigned m2 = (unsigned)__ballot(d2 <= 0.16f);
    if ((m0 >> lane) & 1u) { int p = c0 + __popc(m0 & below); if (p < 16)  o0[p] = j; }
    if ((m1 >> lane) & 1u) { int p = c1 + __popc(m1 & below); if (p < 32)  o1[p] = j; }
    if ((m2 >> lane) & 1u) { int p = c2 + __popc(m2 & below); if (p < 128) o2[p] = j; }
    if (c0 == 0 && m0) f0 = base + __ffs(m0) - 1;
    if (c1 == 0 && m1) f1 = base + __ffs(m1) - 1;
    if (c2 == 0 && m2) f2 = base + __ffs(m2) - 1;
    c0 += __popc(m0); if (c0 > 16)  c0 = 16;
    c1 += __popc(m1); if (c1 > 32)  c1 = 32;
    c2 += __popc(m2); if (c2 > 128) c2 = 128;
    if (c0 == 16 && c1 == 32 && c2 == 128) break;
  }
  for (int p = c0 + lane; p < 16;  p += 32) o0[p] = f0;
  for (int p = c1 + lane; p < 32;  p += 32) o1[p] = f1;
  for (int p = c2 + lane; p < 128; p += 32) o2[p] = f2;
}

// ---------------------------------------------------------------------------
// Kernel 3: weight prep. Convert all 9 layers' W (oc,ic) f32 -> padded f16
// (oc,icp) row-major; prefold BN into (sc, sh) pairs. Single tiny block.
// ---------------------------------------------------------------------------
struct PrepParams {
  const float* W[9];
  const float* b[9];
  const float* g[9];
  const float* be[9];
};

__global__ __launch_bounds__(256) void prep_kernel(PrepParams P,
                                                   _Float16* __restrict__ wf16,
                                                   float* __restrict__ scsh) {
  const int OC[9]  = {32, 32, 64, 64, 64, 128, 64, 96, 128};
  const int IC[9]  = {67, 32, 32, 67, 64, 64,  67, 64, 96};
  const int ICP[9] = {96, 32, 32, 96, 64, 64,  96, 64, 96};
  const float inv_std = rsqrtf(1.0f + 1e-5f);
  int woff = 0, soff = 0;
  for (int l = 0; l < 9; ++l) {
    const int oc = OC[l], ic = IC[l], icp = ICP[l];
    const float* W = P.W[l];
    for (int t = threadIdx.x; t < oc * icp; t += blockDim.x) {
      const int n = t / icp;
      const int c = t - n * icp;
      const float v = (c < ic) ? W[(size_t)n * ic + c] : 0.0f;
      wf16[woff + t] = (_Float16)v;
    }
    for (int n = threadIdx.x; n < oc; n += blockDim.x) {
      const float sc = inv_std * P.g[l][n];
      scsh[soff + 2 * n]     = sc;
      scsh[soff + 2 * n + 1] = P.b[l][n] * sc + P.be[l][n];
    }
    woff += oc * icp;
    soff += 2 * oc;
  }
}

// ---------------------------------------------------------------------------
// Fragment loader (ISA 7.12.2, wave32). Row-major f16 matrix with leading dim
// `ld` (multiple of 32). Each fragment half (8 halves) is contiguous:
//   h[0..7]  = row[kbase + hsel*8 .. +7],  h[8..15] = row[kbase+16+hsel*8 .. +7]
// -> exactly two b128 loads. Works for both A (row=M) and B (row=N) operands.
// ---------------------------------------------------------------------------
__device__ inline v16h load_frag16(const _Float16* __restrict__ p, int ld,
                                   int rbase, int kbase, int lane) {
  const _Float16* rowp =
      p + (size_t)(rbase + (lane & 15)) * ld + kbase + ((lane >> 4) << 3);
  V16H u;
  u.half8[0] = *(const v8h*)(rowp);
  u.half8[1] = *(const v8h*)(rowp + 16);
  return u.v;
}

// One MLP layer: [K x ICP] f16 (LDS) x [OC x ICP]^T f16 (LDS) -> [K x OC] f16
template <int K, int ICP, int OC>
__device__ inline void mlp_layer(const _Float16* __restrict__ xin,
                                 _Float16* __restrict__ xout,
                                 const _Float16* __restrict__ wl,
                                 const float* __restrict__ ss,
                                 int lane, int wv) {
  constexpr int Mt = K / 16, Nt = OC / 16, TILES = Mt * Nt;
  for (int t = wv; t < TILES; t += 4) {    // wave-uniform: EXEC all-1s at WMMA
    const int mt = t / Nt;                 // Nt is constexpr -> no div expansion
    const int nt = t - mt * Nt;
    v8f acc = {};
#pragma unroll
    for (int kc = 0; kc < ICP; kc += 32) {
      v16h a = load_frag16(xin, ICP, mt * 16, kc, lane);
      v16h b = load_frag16(wl,  ICP, nt * 16, kc, lane);
      acc = __builtin_amdgcn_wmma_f32_16x16x32_f16(
          false, a, false, b, (short)0, acc, false, false);
    }
    const int n = nt * 16 + (lane & 15);
    const float sc = ss[2 * n];
    const float sh = ss[2 * n + 1];
#pragma unroll
    for (int r = 0; r < 8; ++r) {
      const int m = mt * 16 + r + ((lane >> 4) << 3);   // C/D layout
      const float y = fmaxf(acc[r] * sc + sh, 0.0f);
      xout[(size_t)m * OC + n] = (_Float16)y;
    }
  }
}

// ---------------------------------------------------------------------------
// Kernel 4: per-centroid gather + 3-layer WMMA MLP + maxpool.
// Weights staged to LDS once (async global->LDS if toolchain exposes it).
// ---------------------------------------------------------------------------
template <int K, int OC0, int OC1, int OC2>
__global__ __launch_bounds__(128) void msg_mlp_kernel(const float* __restrict__ xyz,
                                                      const float* __restrict__ points,
                                                      const float* __restrict__ nxyz,
                                                      const int* __restrict__ gidx,
                                                      const _Float16* __restrict__ wsc,
                                                      const float* __restrict__ ssc,
                                                      float* __restrict__ out_np,
                                                      int ch_base) {
  constexpr int WTOT = OC0 * 96 + OC1 * OC0 + OC2 * OC1;   // halves
  constexpr int W16  = WTOT / 8;                            // 16B chunks
  __shared__ _Float16 xbuf0[K * 128];
  __shared__ _Float16 xbuf1[K * 128];
  __shared__ _Float16 wlds[WTOT];

  const int cid = blockIdx.x;
  const int b = cid >> 9;
  const int s = cid & 511;
  const int tid = threadIdx.x;
  const int lane = tid & 31;
  const int wv = tid >> 5;

  // ---- Stage all 3 layers' f16 weights into LDS (contiguous copy) ----
#if HAVE_ASYNC_LDS
  {
    // Signature per hipcc diagnostic: (int4 AS1*, int4 AS3*, imm off, imm cpol)
    __attribute__((address_space(1))) i4v* gsrc =
        (__attribute__((address_space(1))) i4v*)wsc;
    __attribute__((address_space(3))) i4v* ldst =
        (__attribute__((address_space(3))) i4v*)wlds;
    for (int t = tid; t < W16; t += 128) {
      __builtin_amdgcn_global_load_async_to_lds_b128(gsrc + t, ldst + t, 0, 0);
    }
    __builtin_amdgcn_s_wait_asynccnt(0);
  }
#else
  {
    const uint4* src = (const uint4*)wsc;
    uint4* dst = (uint4*)wlds;
    for (int t = tid; t < W16; t += 128) dst[t] = src[t];
  }
#endif

  // ---- Gather [g_pts | g_xyz - centroid] -> f16 LDS (K x 96), b32-packed ----
  const int* idx = gidx + (size_t)cid * K;
  const float cx = nxyz[(size_t)cid * 3 + 0];
  const float cy = nxyz[(size_t)cid * 3 + 1];
  const float cz = nxyz[(size_t)cid * 3 + 2];
  unsigned* xb32 = (unsigned*)xbuf0;
  for (int t = tid; t < K * 48; t += 128) {
    const int k = t / 48;
    const int cp = t - k * 48;
    const int c = cp * 2;
    const int p = idx[k];
    float v0 = 0.0f, v1 = 0.0f;
    if (c < 64) {
      v0 = points[((size_t)b * 64 + c) * NPTS + p];
      v1 = points[((size_t)b * 64 + c + 1) * NPTS + p];
    } else if (c == 64) {
      v0 = xyz[((size_t)b * 3 + 0) * NPTS + p] - cx;
      v1 = xyz[((size_t)b * 3 + 1) * NPTS + p] - cy;
    } else if (c == 66) {
      v0 = xyz[((size_t)b * 3 + 2) * NPTS + p] - cz;
      v1 = 0.0f;
    }
    H2 pk;
    pk.h[0] = (_Float16)v0;
    pk.h[1] = (_Float16)v1;
    xb32[(size_t)k * 48 + cp] = pk.u;
  }
  __syncthreads();

  // ---- 3 layers, ping-pong buffers, all dims compile-time ----
  mlp_layer<K, 96,  OC0>(xbuf0, xbuf1, wlds,                       ssc,                 lane, wv);
  __syncthreads();
  mlp_layer<K, OC0, OC1>(xbuf1, xbuf0, wlds + OC0 * 96,            ssc + 2 * OC0,       lane, wv);
  __syncthreads();
  mlp_layer<K, OC1, OC2>(xbuf0, xbuf1, wlds + OC0 * 96 + OC1*OC0,  ssc + 2 * (OC0+OC1), lane, wv);
  __syncthreads();

  // ---- Max over K neighbors (post-ReLU => >= 0) ----
  for (int n = tid; n < OC2; n += 128) {
    float mx = 0.0f;
#pragma unroll 4
    for (int k = 0; k < K; ++k)
      mx = fmaxf(mx, (float)xbuf1[(size_t)k * OC2 + n]);
    out_np[((size_t)b * 320 + ch_base + n) * NPT + s] = mx;
  }
}

// ---------------------------------------------------------------------------
extern "C" void kernel_launch(void* const* d_in, const int* in_sizes, int n_in,
                              void* d_out, int out_size, void* d_ws, size_t ws_size,
                              hipStream_t stream) {
  const float* xyz    = (const float*)d_in[0];
  const float* points = (const float*)d_in[1];
  auto getp = [&](int sc, int l, int which) -> const float* {
    return (const float*)d_in[2 + (sc * 3 + l) * 4 + which];
  };

  // Workspace layout: new_xyz | idx0 | idx1 | idx2 | wf16 | scsh
  float* ws_nxyz = (float*)d_ws;
  int* idx0 = (int*)(ws_nxyz + (size_t)BATCH * NPT * 3);
  int* idx1 = idx0 + (size_t)BATCH * NPT * 16;
  int* idx2 = idx1 + (size_t)BATCH * NPT * 32;
  _Float16* wf16 = (_Float16*)(idx2 + (size_t)BATCH * NPT * 128);
  // total f16 weights: 6144 + 18432 + 24576 = 49152 halves
  float* scsh = (float*)(wf16 + 49152);

  float* out = (float*)d_out;                       // new_xyz (B,3,512)
  float* out_np = out + (size_t)BATCH * 3 * NPT;    // new_points (B,320,512)

  PrepParams PP;
  for (int sc = 0; sc < 3; ++sc)
    for (int l = 0; l < 3; ++l) {
      PP.W[sc * 3 + l]  = getp(sc, l, 0);
      PP.b[sc * 3 + l]  = getp(sc, l, 1);
      PP.g[sc * 3 + l]  = getp(sc, l, 2);
      PP.be[sc * 3 + l] = getp(sc, l, 3);
    }

  fps_kernel<<<BATCH, 256, 0, stream>>>(xyz, ws_nxyz, out);
  prep_kernel<<<1, 256, 0, stream>>>(PP, wf16, scsh);
  ballquery_kernel<<<(BATCH * NPT) / 8, 256, 0, stream>>>(xyz, ws_nxyz, idx0, idx1, idx2);

  const int ncent = BATCH * NPT;
  // weight / scsh offsets per scale (halves / floats)
  msg_mlp_kernel<16, 32, 32, 64>
      <<<ncent, 128, 0, stream>>>(xyz, points, ws_nxyz, idx0, wf16,         scsh,       out_np, 0);
  msg_mlp_kernel<32, 64, 64, 128>
      <<<ncent, 128, 0, stream>>>(xyz, points, ws_nxyz, idx1, wf16 + 6144,  scsh + 256, out_np, 64);
  msg_mlp_kernel<128, 64, 96, 128>
      <<<ncent, 128, 0, stream>>>(xyz, points, ws_nxyz, idx2, wf16 + 24576, scsh + 768, out_np, 192);
}
